// MHSA_Intra_69380901700004
// MI455X (gfx1250) — matrix operand order repaired and checked by reading, not compiled, not run on hardware
//
#include <hip/hip_runtime.h>
#include <hip/hip_bf16.h>

// Problem constants (match reference)
#define Bsz 2
#define Cdim 1024
#define Hn 16
#define DHd 64
#define Tn 2048

typedef __attribute__((ext_vector_type(16))) __bf16 v16bf;
typedef __attribute__((ext_vector_type(8)))  float  v8f;
typedef int v4i __attribute__((__vector_size__(16)));  // matches builtin param

// ---- CDNA5 feature detection (compile-safe on host pass and older tools) ----
#if defined(__gfx1250__) && __has_builtin(__builtin_amdgcn_global_load_async_to_lds_b128)
#define HAVE_ASYNC 1
#else
#define HAVE_ASYNC 0
#endif
#if defined(__gfx1250__) && __has_builtin(__builtin_amdgcn_tensor_load_to_lds)
#define HAVE_TDM 1
#else
#define HAVE_TDM 0
#endif

static __device__ inline void wait_asynccnt0() {
#if __has_builtin(__builtin_amdgcn_s_wait_asynccnt)
  __builtin_amdgcn_s_wait_asynccnt(0);
#else
  asm volatile("s_wait_asynccnt 0x0" ::: "memory");
#endif
}

static __device__ inline void wait_tensorcnt0() {
#if __has_builtin(__builtin_amdgcn_s_wait_tensorcnt)
  __builtin_amdgcn_s_wait_tensorcnt(0);
#else
  asm volatile("s_wait_tensorcnt 0x0" ::: "memory");
#endif
}

#if HAVE_TDM
typedef unsigned int u32x4 __attribute__((ext_vector_type(4)));
typedef int          i32x4 __attribute__((ext_vector_type(4)));
typedef int          i32x8 __attribute__((ext_vector_type(8)));

// TDM load of a 2D bf16 tile: rows x cols elements, global row pitch in
// elements, into LDS at lds_byte_off with rows padded 64 elems -> 72 elems
// (pad_amount = 4 DWORDs every pad_interval = 32 DWORDs).  D# layout per
// CDNA5 ISA section 8.3 - 8.6.
static __device__ inline void tdm_load_tile_bf16(const __bf16* gsrc,
                                                 unsigned lds_byte_off,
                                                 unsigned rows, unsigned cols,
                                                 unsigned pitch_elems) {
  const unsigned long long ga = (unsigned long long)(uintptr_t)gsrc;
  u32x4 g0;
  g0[0] = 1u;                                   // count=1 (valid user D#)
  g0[1] = lds_byte_off;                         // lds_addr
  g0[2] = (unsigned)(ga & 0xFFFFFFFFu);         // global_addr lo
  g0[3] = (unsigned)((ga >> 32) & 0x01FFFFFFu)  // global_addr hi (bits 56:32)
          | (2u << 30);                         // type = 2 ("image")
  i32x8 g1;
  g1[0] = (1 << 16)                             // data_size = 2 bytes
        | (1 << 20)                             // pad_enable
        | (4 << 22)                             // pad_interval: 32 DWORDs
        | (3 << 25);                            // pad_amount : 4 DWORDs
  g1[1] = (int)((cols & 0xFFFFu) << 16);        // tensor_dim0[15:0]
  g1[2] = (int)((cols >> 16) | ((rows & 0xFFFFu) << 16)); // dim0 hi | dim1 lo
  g1[3] = (int)((rows >> 16) | ((cols & 0xFFFFu) << 16)); // dim1 hi | tile_dim0
  g1[4] = (int)(rows & 0xFFFFu);                // tile_dim1 (tile_dim2 = 0)
  g1[5] = (int)pitch_elems;                     // tensor_dim0_stride lo
  g1[6] = 0;
  g1[7] = 0;
  i32x4 z4; z4[0] = z4[1] = z4[2] = z4[3] = 0;
#if __clang_major__ >= 23
  i32x8 z8;
#pragma unroll
  for (int i = 0; i < 8; ++i) z8[i] = 0;
  __builtin_amdgcn_tensor_load_to_lds(g0, g1, z4, z4, z8, 0);
#else
  __builtin_amdgcn_tensor_load_to_lds(g0, g1, z4, z4, 0);
#endif
}
#endif  // HAVE_TDM

static __device__ inline unsigned lds_off_u32(const void* p) {
  return (unsigned)(uintptr_t)p;  // low 32 bits of generic addr = LDS offset
}

// ---------------------------------------------------------------------------
// WMMA fragment loaders (CDNA5 ISA 7.12.2 layouts), bf16 16x16x32.
// A (16x32): lane L = g*16+m holds row m; VGPR j<4: K = g*8+2j(+1);
//            VGPR j>=4: K = 16+g*8+2(j-4)(+1).  LDS tile [m][k], k-contig.
// B (32x16): lane L = g*16+n holds col n; same K mapping as A.
//            LDS tile stored as [n][k], k-contig.
// C/D (16x16 f32): lane L = g*16+n; VGPR r holds row (r + 8g), col n.
// ---------------------------------------------------------------------------
static __device__ inline v16bf load_a_frag(const __bf16* sm, int stride,
                                           int mBase, int kOff, int lane) {
  const int m = lane & 15, g = lane >> 4;
  const __bf16* p = sm + (size_t)(mBase + m) * stride + kOff;
  v16bf a;
#pragma unroll
  for (int j = 0; j < 8; ++j) {
    const int kb = ((j < 4) ? 0 : 16) + g * 8 + 2 * (j & 3);
    a[2 * j]     = p[kb];
    a[2 * j + 1] = p[kb + 1];
  }
  return a;
}

static __device__ inline v16bf load_b_frag(const __bf16* sm, int stride,
                                           int nBase, int kOff, int lane) {
  const int n = lane & 15, g = lane >> 4;
  const __bf16* p = sm + (size_t)(nBase + n) * stride + kOff;
  v16bf b;
#pragma unroll
  for (int j = 0; j < 8; ++j) {
    const int kb = ((j < 4) ? 0 : 16) + g * 8 + 2 * (j & 3);
    b[2 * j]     = p[kb];
    b[2 * j + 1] = p[kb + 1];
  }
  return b;
}

// 16-byte tile-row copy unit: global -> LDS (async if available)
static __device__ inline void cp16(const __bf16* g, __bf16* l) {
#if HAVE_ASYNC
  __builtin_amdgcn_global_load_async_to_lds_b128((v4i*)g, (v4i*)l, 0, 0);
#else
  *(uint4*)l = *(const uint4*)g;
#endif
}

// ---------------------------------------------------------------------------
// f32 -> bf16 elementwise convert (weights)
// ---------------------------------------------------------------------------
__global__ __launch_bounds__(256) void cvt_f32_bf16(const float* __restrict__ src,
                                                    __bf16* __restrict__ dst, int n) {
  int i = blockIdx.x * 256 + threadIdx.x;
  if (i < n) dst[i] = (__bf16)src[i];
}

// ---------------------------------------------------------------------------
// Fused transpose + convert: x [B,C,T] f32 -> xT [B,T,C] bf16
// ---------------------------------------------------------------------------
__global__ __launch_bounds__(256) void transpose_cvt(const float* __restrict__ x,
                                                     __bf16* __restrict__ xT) {
  __shared__ float tile[32][33];
  const int t0 = blockIdx.x * 32, c0 = blockIdx.y * 32, b = blockIdx.z;
  const int tx = threadIdx.x & 31, ty = threadIdx.x >> 5;  // 8 rows of 32
#pragma unroll
  for (int i = 0; i < 4; ++i)
    tile[ty + 8 * i][tx] = x[((size_t)b * Cdim + c0 + ty + 8 * i) * Tn + t0 + tx];
  __syncthreads();
#pragma unroll
  for (int i = 0; i < 4; ++i)
    xT[((size_t)b * Tn + t0 + ty + 8 * i) * Cdim + c0 + tx] =
        (__bf16)tile[tx][ty + 8 * i];
}

// ---------------------------------------------------------------------------
// Projection GEMM: D[M=C, N=T] = W[M,K=C] * XT[b][N,K]^T + bias
// XT is [B,T,C] bf16 (k-contiguous rows) so both LDS tiles are row copies.
// Block = 128 threads (4 waves), 64x64 output tile, K staged 32 at a time.
// mode 0: bf16 [B,H,T,DH] * scale   (Q with scale=1/8, K with scale=1)
// mode 1: bf16 [B,C,T]              (V -> [B,H,DH,T])
// mode 3: f32  [B,C,T]              (final Wo projection -> y)
// ---------------------------------------------------------------------------
__global__ __launch_bounds__(128) void wmma_gemm_proj(
    const __bf16* __restrict__ W, const __bf16* __restrict__ XT,
    const float* __restrict__ bias, void* __restrict__ dst,
    int mode, float scale) {
  __shared__ __bf16 As[64 * 40];  // [m][k], stride 40
  __shared__ __bf16 Bs[64 * 40];  // [n][k], stride 40

  const int tid = threadIdx.x;
  const int lane = tid & 31, wid = tid >> 5;
  const int n0 = blockIdx.x * 64;  // t tile
  const int m0 = blockIdx.y * 64;  // d tile
  const int b  = blockIdx.z;
  const __bf16* Xb = XT + (size_t)b * Tn * Cdim;

  v8f acc[4];
#pragma unroll
  for (int s = 0; s < 4; ++s)
#pragma unroll
    for (int i = 0; i < 8; ++i) acc[s][i] = 0.0f;

  for (int k0 = 0; k0 < Cdim; k0 += 32) {
    // stage A (64x32 from W) and B (64x32 from XT): pure row copies
#pragma unroll
    for (int i = 0; i < 2; ++i) {
      const int u = tid + 128 * i;     // 0..255 : 64 rows x 4 octets
      const int r = u >> 2, oc = (u & 3) * 8;
      cp16(W  + (size_t)(m0 + r) * Cdim + k0 + oc, As + r * 40 + oc);
      cp16(Xb + (size_t)(n0 + r) * Cdim + k0 + oc, Bs + r * 40 + oc);
    }
#if HAVE_ASYNC
    wait_asynccnt0();
#endif
    if (k0 + 32 < Cdim)
      __builtin_prefetch(W + (size_t)(m0 + (tid >> 2)) * Cdim + k0 + 32, 0, 1);
    __syncthreads();

    v16bf a = load_a_frag(As, 40, wid * 16, 0, lane);
#pragma unroll
    for (int s = 0; s < 4; ++s) {
      v16bf bf = load_b_frag(Bs, 40, s * 16, 0, lane);
      acc[s] = __builtin_amdgcn_wmma_f32_16x16x32_bf16(
          false, a, false, bf, (short)0, acc[s], false, false);
    }
    __syncthreads();
  }

  // epilogue
  const int nl = lane & 15, g = lane >> 4;
#pragma unroll
  for (int s = 0; s < 4; ++s) {
    const int t = n0 + s * 16 + nl;
#pragma unroll
    for (int r = 0; r < 8; ++r) {
      const int d = m0 + wid * 16 + r + 8 * g;
      const float val = acc[s][r] + bias[d];
      if (mode == 3) {
        ((float*)dst)[((size_t)b * Cdim + d) * Tn + t] = val;
      } else if (mode == 1) {
        ((__bf16*)dst)[((size_t)b * Cdim + d) * Tn + t] = (__bf16)val;
      } else {
        const int h = d >> 6, dh = d & 63;
        ((__bf16*)dst)[(((size_t)b * Hn + h) * Tn + t) * DHd + dh] =
            (__bf16)(val * scale);
      }
    }
  }
}

// ---------------------------------------------------------------------------
// Flash attention: per (b, h, 64-row t-tile). 4 waves, each owns 16 t-rows.
// Q: [B,H,T,DH] bf16 (pre-scaled)   K: [B,H,T,DH] bf16   V: [B,C,T] bf16
// O: [B,T,C] bf16 (c-contiguous, feeds Wo GEMM directly)
// All three LDS tiles are rectangular row copies -> TDM-eligible.
// ---------------------------------------------------------------------------
__global__ __launch_bounds__(128) void wmma_attention(
    const __bf16* __restrict__ Q, const __bf16* __restrict__ K,
    const __bf16* __restrict__ V, __bf16* __restrict__ O) {
  __shared__ __bf16 Qs[64 * 72];  // [t][dh]
  __shared__ __bf16 Ks[64 * 72];  // [s][dh]
  __shared__ __bf16 Vs[64 * 72];  // [dh][s]
  __shared__ __bf16 Ps[64 * 72];  // [t][s] softmax probs

  const int tid = threadIdx.x, lane = tid & 31, wid = tid >> 5;
  const int t0 = blockIdx.x * 64;
  const int h = blockIdx.y, b = blockIdx.z;
  const __bf16* Qbh = Q + ((size_t)b * Hn + h) * Tn * DHd;
  const __bf16* Kbh = K + ((size_t)b * Hn + h) * Tn * DHd;
  const __bf16* Vbh = V + ((size_t)b * Cdim + h * DHd) * Tn;

  // ---- stage Q tile once ----
#if HAVE_TDM
  if (wid == 0) {
    tdm_load_tile_bf16(Qbh + (size_t)t0 * DHd, lds_off_u32(Qs), 64, 64, 64);
    wait_tensorcnt0();
  }
#else
#pragma unroll
  for (int u0 = 0; u0 < 4; ++u0) {
    const int u = tid + 128 * u0;
    const int r = u >> 3, oc = (u & 7) * 8;
    cp16(Qbh + (size_t)(t0 + r) * DHd + oc, Qs + r * 72 + oc);
  }
#if HAVE_ASYNC
  wait_asynccnt0();
#endif
#endif
  __syncthreads();

  const v16bf aq0 = load_a_frag(Qs, 72, wid * 16, 0, lane);
  const v16bf aq1 = load_a_frag(Qs, 72, wid * 16, 32, lane);

  float m[8], l[8];
#pragma unroll
  for (int r = 0; r < 8; ++r) { m[r] = -1e30f; l[r] = 0.0f; }
  v8f accO[4];
#pragma unroll
  for (int s = 0; s < 4; ++s)
#pragma unroll
    for (int i = 0; i < 8; ++i) accO[s][i] = 0.0f;

  for (int s0 = 0; s0 < Tn; s0 += 64) {
    // ---- stage K tile [s][dh] and V tile [dh][s] ----
#if HAVE_TDM
    if (wid == 0) {
      tdm_load_tile_bf16(Kbh + (size_t)s0 * DHd, lds_off_u32(Ks), 64, 64, 64);
      tdm_load_tile_bf16(Vbh + s0, lds_off_u32(Vs), 64, 64, Tn);
      wait_tensorcnt0();
    }
#else
#pragma unroll
    for (int u0 = 0; u0 < 4; ++u0) {
      const int u = tid + 128 * u0;
      const int r = u >> 3, oc = (u & 7) * 8;
      cp16(Kbh + (size_t)(s0 + r) * DHd + oc, Ks + r * 72 + oc);
      cp16(Vbh + (size_t)r * Tn + s0 + oc,    Vs + r * 72 + oc);
    }
#if HAVE_ASYNC
    wait_asynccnt0();
#endif
#endif
    __syncthreads();

    // ---- S = Q * K^T (16t x 64s per wave, K-dim = 64) ----
    v8f accS[4];
#pragma unroll
    for (int s = 0; s < 4; ++s)
#pragma unroll
      for (int i = 0; i < 8; ++i) accS[s][i] = 0.0f;
#pragma unroll
    for (int sub = 0; sub < 4; ++sub) {
      v16bf b0 = load_b_frag(Ks, 72, sub * 16, 0, lane);
      accS[sub] = __builtin_amdgcn_wmma_f32_16x16x32_bf16(
          false, aq0, false, b0, (short)0, accS[sub], false, false);
      v16bf b1 = load_b_frag(Ks, 72, sub * 16, 32, lane);
      accS[sub] = __builtin_amdgcn_wmma_f32_16x16x32_bf16(
          false, aq1, false, b1, (short)0, accS[sub], false, false);
    }

    // ---- online softmax; row (wid*16 + r + 8g) lives in 16-lane half g ----
#pragma unroll
    for (int r = 0; r < 8; ++r) {
      float rm = fmaxf(fmaxf(accS[0][r], accS[1][r]),
                       fmaxf(accS[2][r], accS[3][r]));
#pragma unroll
      for (int off = 1; off < 16; off <<= 1) rm = fmaxf(rm, __shfl_xor(rm, off));
      const float mn = fmaxf(m[r], rm);
      const float alpha = __expf(m[r] - mn);
      float rs = 0.0f;
#pragma unroll
      for (int sub = 0; sub < 4; ++sub) {
        const float p = __expf(accS[sub][r] - mn);
        accS[sub][r] = p;
        rs += p;
      }
#pragma unroll
      for (int off = 1; off < 16; off <<= 1) rs += __shfl_xor(rs, off);
      l[r] = l[r] * alpha + rs;
      m[r] = mn;
#pragma unroll
      for (int sub = 0; sub < 4; ++sub) accO[sub][r] *= alpha;
    }

    // ---- spill P tile to LDS as bf16 (C/D layout -> [t][s]) ----
    {
      const int n = lane & 15, g = lane >> 4;
#pragma unroll
      for (int sub = 0; sub < 4; ++sub)
#pragma unroll
        for (int r = 0; r < 8; ++r)
          Ps[(wid * 16 + r + 8 * g) * 72 + sub * 16 + n] = (__bf16)accS[sub][r];
    }
    __syncthreads();

    // ---- O += P * V (K-dim = 64 over s) ----
    const v16bf ap0 = load_a_frag(Ps, 72, wid * 16, 0, lane);
    const v16bf ap1 = load_a_frag(Ps, 72, wid * 16, 32, lane);
#pragma unroll
    for (int sub = 0; sub < 4; ++sub) {
      v16bf b0 = load_b_frag(Vs, 72, sub * 16, 0, lane);
      accO[sub] = __builtin_amdgcn_wmma_f32_16x16x32_bf16(
          false, ap0, false, b0, (short)0, accO[sub], false, false);
      v16bf b1 = load_b_frag(Vs, 72, sub * 16, 32, lane);
      accO[sub] = __builtin_amdgcn_wmma_f32_16x16x32_bf16(
          false, ap1, false, b1, (short)0, accO[sub], false, false);
    }
    __syncthreads();  // protect Ks/Vs/Ps before next iteration's staging
  }

  // ---- epilogue: normalize, write O[b, t, c=h*64+dh] ----
  const int n = lane & 15, g = lane >> 4;
#pragma unroll
  for (int sub = 0; sub < 4; ++sub) {
    const int dh = sub * 16 + n;
#pragma unroll
    for (int r = 0; r < 8; ++r) {
      const int t = t0 + wid * 16 + r + 8 * g;
      O[((size_t)b * Tn + t) * Cdim + h * DHd + dh] =
          (__bf16)(accO[sub][r] / l[r]);
    }
  }
}

// ---------------------------------------------------------------------------
// BatchNorm1d (batch stats over B,T) + affine + residual.  One block/channel.
// ---------------------------------------------------------------------------
__global__ __launch_bounds__(256) void bn_residual(
    const float* __restrict__ y, const float* __restrict__ x,
    const float* __restrict__ gamma, const float* __restrict__ beta,
    float* __restrict__ out) {
  const int c = blockIdx.x;
  const int tid = threadIdx.x;
  const int lane = tid & 31, wid = tid >> 5;

  float s = 0.0f, sq = 0.0f;
  for (int b = 0; b < Bsz; ++b) {
    const float* yr = y + ((size_t)b * Cdim + c) * Tn;
    for (int t = tid; t < Tn; t += 256) {
      const float v = yr[t];
      s += v;
      sq += v * v;
    }
  }
#pragma unroll
  for (int off = 1; off < 32; off <<= 1) {
    s += __shfl_xor(s, off);
    sq += __shfl_xor(sq, off);
  }
  __shared__ float rs[2][8];
  if (lane == 0) { rs[0][wid] = s; rs[1][wid] = sq; }
  __syncthreads();
  if (wid == 0) {
    s = (lane < 8) ? rs[0][lane] : 0.0f;
    sq = (lane < 8) ? rs[1][lane] : 0.0f;
#pragma unroll
    for (int off = 1; off < 8; off <<= 1) {
      s += __shfl_xor(s, off);
      sq += __shfl_xor(sq, off);
    }
    if (lane == 0) { rs[0][0] = s; rs[1][0] = sq; }
  }
  __syncthreads();

  const float inv_n = 1.0f / (float)(Bsz * Tn);
  const float mean = rs[0][0] * inv_n;
  const float var = rs[1][0] * inv_n - mean * mean;
  const float rstd = rsqrtf(var + 1e-5f);
  const float gm = gamma[c] * rstd, bt = beta[c];

  for (int b = 0; b < Bsz; ++b) {
    const size_t base = ((size_t)b * Cdim + c) * Tn;
    for (int t = tid; t < Tn; t += 256)
      out[base + t] = x[base + t] + (y[base + t] - mean) * gm + bt;
  }
}

// ---------------------------------------------------------------------------
// Host-side orchestration
// ---------------------------------------------------------------------------
extern "C" void kernel_launch(void* const* d_in, const int* in_sizes, int n_in,
                              void* d_out, int out_size, void* d_ws, size_t ws_size,
                              hipStream_t stream) {
  (void)in_sizes; (void)n_in; (void)out_size; (void)ws_size;
  const float* x     = (const float*)d_in[0];
  const float* Wq    = (const float*)d_in[1];
  const float* bq    = (const float*)d_in[2];
  const float* Wk    = (const float*)d_in[3];
  const float* bk    = (const float*)d_in[4];
  const float* Wv    = (const float*)d_in[5];
  const float* bv    = (const float*)d_in[6];
  const float* Wo    = (const float*)d_in[7];
  const float* bo    = (const float*)d_in[8];
  const float* gamma = (const float*)d_in[9];
  const float* beta  = (const float*)d_in[10];

  const size_t NX = (size_t)Bsz * Cdim * Tn;  // 4M elems
  const size_t NW = (size_t)Cdim * Cdim;      // 1M elems

  char* ws = (char*)d_ws;
  size_t off = 0;
  __bf16* xT   = (__bf16*)(ws + off); off += NX * 2;  // [B,T,C]
  __bf16* Wqb  = (__bf16*)(ws + off); off += NW * 2;
  __bf16* Wkb  = (__bf16*)(ws + off); off += NW * 2;
  __bf16* Wvb  = (__bf16*)(ws + off); off += NW * 2;
  __bf16* Wob  = (__bf16*)(ws + off); off += NW * 2;
  __bf16* qbf  = (__bf16*)(ws + off); off += NX * 2;  // [B,H,T,DH]
  __bf16* kbf  = (__bf16*)(ws + off); off += NX * 2;  // [B,H,T,DH]
  __bf16* vbf  = (__bf16*)(ws + off); off += NX * 2;  // [B,C,T] = [B,H,DH,T]
  __bf16* obf  = (__bf16*)(ws + off); off += NX * 2;  // [B,T,C]
  float*  ybuf = (float*) (ws + off); off += NX * 4;  // [B,C,T] f32

  // input transpose+convert, weight converts
  transpose_cvt<<<dim3(Tn / 32, Cdim / 32, Bsz), dim3(256), 0, stream>>>(x, xT);
  cvt_f32_bf16<<<dim3((unsigned)((NW + 255) / 256)), dim3(256), 0, stream>>>(Wq, Wqb, (int)NW);
  cvt_f32_bf16<<<dim3((unsigned)((NW + 255) / 256)), dim3(256), 0, stream>>>(Wk, Wkb, (int)NW);
  cvt_f32_bf16<<<dim3((unsigned)((NW + 255) / 256)), dim3(256), 0, stream>>>(Wv, Wvb, (int)NW);
  cvt_f32_bf16<<<dim3((unsigned)((NW + 255) / 256)), dim3(256), 0, stream>>>(Wo, Wob, (int)NW);

  // projections (Q pre-scaled by DH^-0.5)
  const dim3 gg(Tn / 64, Cdim / 64, Bsz);
  wmma_gemm_proj<<<gg, dim3(128), 0, stream>>>(Wqb, xT, bq, (void*)qbf, 0, 0.125f);
  wmma_gemm_proj<<<gg, dim3(128), 0, stream>>>(Wkb, xT, bk, (void*)kbf, 0, 1.0f);
  wmma_gemm_proj<<<gg, dim3(128), 0, stream>>>(Wvb, xT, bv, (void*)vbf, 1, 1.0f);

  // flash attention
  wmma_attention<<<dim3(Tn / 64, Hn, Bsz), dim3(128), 0, stream>>>(qbf, kbf, vbf, obf);

  // output projection -> y (f32, [B,C,T])
  wmma_gemm_proj<<<gg, dim3(128), 0, stream>>>(Wob, obf, bo, (void*)ybuf, 3, 1.0f);

  // batchnorm + residual -> out
  bn_residual<<<dim3(Cdim), dim3(256), 0, stream>>>(ybuf, x, gamma, beta, (float*)d_out);
}
